// MultiHeadSelfAttention_16973710753929
// MI455X (gfx1250) — matrix operand order repaired and checked by reading, not compiled, not run on hardware
//
#include <hip/hip_runtime.h>

// ---------------------------------------------------------------------------
// MultiHeadSelfAttention for MI455X (gfx1250, wave32, WMMA)
//   x:[4,2048,1024] f32 -> out:[4,2048,1024] f32
//   bf16 WMMA (16x16x32, f32 accum) for all three matmul stages,
//   fp32 streaming (flash) softmax, double-buffered async K/V LDS staging.
// ---------------------------------------------------------------------------

typedef __bf16 bf16_t;
typedef __attribute__((ext_vector_type(16))) __bf16 v16bf;
typedef __attribute__((ext_vector_type(8)))  float  v8f;
typedef __attribute__((ext_vector_type(4)))  unsigned int u32x4;

#define EMBED 1024
#define HEADS 16
#define DHEAD 64
#define SLEN  2048
#define BATCH 4

union BF16Frag { v16bf v; u32x4 q[2]; };

// A-fragment (16x32, bf16) from row-major [.., ld] storage (global or LDS).
// Lane m = lane&15 holds K chunks [koff..koff+7] and [koff+16..koff+23],
// koff = 0 for lanes 0-15, 8 for lanes 16-31 (per CDNA5 ISA 7.12.2).
__device__ __forceinline__ v16bf load_a_frag(const bf16_t* base, int ld,
                                             int row0, int k0) {
  const int lane = threadIdx.x & 31;
  const int m    = lane & 15;
  const int koff = (lane < 16) ? 0 : 8;
  const bf16_t* p = base + (size_t)(row0 + m) * ld + (k0 + koff);
  BF16Frag f;
  f.q[0] = *(const u32x4*)(p);
  f.q[1] = *(const u32x4*)(p + 16);
  return f.v;
}

// B-fragment (32x16, bf16) from TRANSPOSED storage Bt[n][k] (row-major, ld=K).
// Lane n = lane&15 holds K chunks [koff..koff+7] and [koff+16..koff+23], koff = 0/8 per lane half.
__device__ __forceinline__ v16bf load_b_frag_t(const bf16_t* baseT, int ld,
                                               int n0, int k0) {
  const int lane = threadIdx.x & 31;
  const int n    = lane & 15;
  const int ks   = k0 + ((lane < 16) ? 0 : 8);
  const bf16_t* p = baseT + (size_t)(n0 + n) * ld + ks;
  BF16Frag f;
  f.q[0] = *(const u32x4*)(p);
  f.q[1] = *(const u32x4*)(p + 16);
  return f.v;
}

__device__ __forceinline__ v8f wmma_bf16(v16bf a, v16bf b, v8f c) {
  return __builtin_amdgcn_wmma_f32_16x16x32_bf16(false, a, false, b,
                                                 (short)0, c, false, false);
}

// Async global -> LDS copy (16 B per lane), tracked by ASYNCcnt.
// VDST operand carries the LDS byte offset = low 32 bits of the generic
// shared-aperture pointer (ISA 10.2).
__device__ __forceinline__ void async_ld_b128(bf16_t* dst_lds,
                                              const bf16_t* src_g) {
  unsigned loff = (unsigned)(size_t)(void*)dst_lds;
  unsigned long long g = (unsigned long long)(size_t)(const void*)src_g;
  asm volatile("global_load_async_to_lds_b128 %0, %1, off"
               :: "v"(loff), "v"(g) : "memory");
}
__device__ __forceinline__ void wait_async0() {
  asm volatile("s_wait_asynccnt 0" ::: "memory");
}

// ---------------------------------------------------------------------------
// Conversions
// ---------------------------------------------------------------------------
__global__ void cvt_bf16_kernel(const float* __restrict__ src,
                                bf16_t* __restrict__ dst, int n) {
  int i = blockIdx.x * blockDim.x + threadIdx.x;
  if (i < n) dst[i] = (bf16_t)src[i];
}

// src [rows][COLS] f32 -> dst [COLS][rows] bf16 (transpose for B-fragments)
template <int COLS>
__global__ void cvt_bf16_tr_kernel(const float* __restrict__ src,
                                   bf16_t* __restrict__ dst, int rows) {
  int i = blockIdx.x * blockDim.x + threadIdx.x;
  if (i < rows * COLS) {
    int r = i / COLS, c = i % COLS;
    dst[(size_t)c * rows + r] = (bf16_t)src[i];
  }
}

// ---------------------------------------------------------------------------
// QKV GEMM: [8192,1024]bf16 @ [1024,3072] (B stored transposed [3072][1024]).
// Wave tile 32x64 (2 A-frags x 4 B-frags -> 8 WMMA per K-step), WG tile 128x128.
// Epilogue scatters into Q[B,H,S,Dh] (pre-scaled by 1/8), K[B,H,S,Dh],
// Vt[B,H,Dh,S].
// ---------------------------------------------------------------------------
__global__ __launch_bounds__(256)
void qkv_gemm_kernel(const bf16_t* __restrict__ Xb, const bf16_t* __restrict__ Wt,
                     const float* __restrict__ bias,
                     bf16_t* __restrict__ Qb, bf16_t* __restrict__ Kb,
                     bf16_t* __restrict__ Vt) {
  const int wave = threadIdx.x >> 5;
  const int lane = threadIdx.x & 31;
  const int M0 = blockIdx.x * 128 + (wave >> 1) * 32;   // 2 x 16-row frags
  const int N0 = blockIdx.y * 128 + (wave & 1) * 64;    // 4 x 16-col frags

  v8f acc[2][4] = {};
  for (int k0 = 0; k0 < EMBED; k0 += 32) {
    v16bf a0 = load_a_frag(Xb, EMBED, M0, k0);
    v16bf a1 = load_a_frag(Xb, EMBED, M0 + 16, k0);
#pragma unroll
    for (int ni = 0; ni < 4; ++ni) {
      v16bf b = load_b_frag_t(Wt, EMBED, N0 + ni * 16, k0);
      acc[0][ni] = wmma_bf16(a0, b, acc[0][ni]);
      acc[1][ni] = wmma_bf16(a1, b, acc[1][ni]);
    }
  }

  const int rowoff = (lane < 16) ? 0 : 8;
#pragma unroll
  for (int ni = 0; ni < 4; ++ni) {
    const int n = N0 + ni * 16 + (lane & 15);
    const float bv = bias[n];
    const int which = n >> 10;      // 0=Q 1=K 2=V
    const int e = n & 1023;
    const int h = e >> 6;
    const int d = e & 63;
#pragma unroll
    for (int mi = 0; mi < 2; ++mi) {
#pragma unroll
      for (int i = 0; i < 8; ++i) {
        const int r = M0 + mi * 16 + rowoff + i;
        const int b = r >> 11;
        const int s = r & 2047;
        const float val = acc[mi][ni][i] + bv;
        const size_t bh = (size_t)(b * HEADS + h);
        if (which == 0)      Qb[(bh * SLEN + s) * DHEAD + d] = (bf16_t)(val * 0.125f);
        else if (which == 1) Kb[(bh * SLEN + s) * DHEAD + d] = (bf16_t)val;
        else                 Vt[(bh * DHEAD + d) * SLEN + s] = (bf16_t)val;
      }
    }
  }
}

// ---------------------------------------------------------------------------
// Flash attention: block = 4 waves = 64 query rows of one (b,h).
// Double-buffered 32-key K/V tiles staged via global_load_async_to_lds_b128:
// tile i+1 streams into LDS (ASYNCcnt) while tile i is consumed by WMMA.
// Scores via 4 WMMA from LDS B-frags, fp32 online softmax with shfl_xor row
// reductions, P tile through per-wave LDS -> A-fragment, 4 WMMA for P@V.
// ---------------------------------------------------------------------------
__global__ __launch_bounds__(128)
void attn_kernel(const bf16_t* __restrict__ Qb, const bf16_t* __restrict__ Kb,
                 const bf16_t* __restrict__ Vt, bf16_t* __restrict__ Ob) {
  __shared__ bf16_t Ktile[2][32 * 64];   // [buf][t_local][d]
  __shared__ bf16_t Vtile[2][64 * 32];   // [buf][d][t_local]
  __shared__ bf16_t plds[4][16 * 32];    // per-wave P tile

  const int wave = threadIdx.x >> 5;
  const int lane = threadIdx.x & 31;
  const int tid  = threadIdx.x;
  const int bh   = blockIdx.x >> 5;          // 64 (b,h) pairs
  const int srow = (blockIdx.x & 31) * 64 + wave * 16;

  const bf16_t* Qp = Qb + (size_t)bh * SLEN * DHEAD;
  const bf16_t* Kp = Kb + (size_t)bh * SLEN * DHEAD;
  const bf16_t* Vp = Vt + (size_t)bh * DHEAD * SLEN;

  // cooperative-copy index decomposition (128 threads, 2 x 16B chunks each)
  const int ktrow0 = tid >> 3,  kdch = (tid & 7) * 8;          // K chunks
  const int ktrow1 = (tid + 128) >> 3;
  const int vdrow0 = tid >> 2,  vtch = (tid & 3) * 8;          // V chunks
  const int vdrow1 = (tid + 128) >> 2;

  // Q fragments (16 rows x 64 d), loaded once. Q already scaled by 1/sqrt(Dh).
  const v16bf aq0 = load_a_frag(Qp, DHEAD, srow, 0);
  const v16bf aq1 = load_a_frag(Qp, DHEAD, srow, 32);

  v8f oc[4] = {};                 // 16x64 output accumulator (4 C tiles)
  float mrow[8], lrow[8];
#pragma unroll
  for (int i = 0; i < 8; ++i) { mrow[i] = -__builtin_inff(); lrow[i] = 0.f; }

  bf16_t* pl = plds[wave];
  const int rowoff = (lane < 16) ? 0 : 8;
  const int ncol   = lane & 15;

  // prologue: kick off tile 0 into buffer 0
  {
    async_ld_b128(&Ktile[0][ktrow0 * 64 + kdch], &Kp[(size_t)ktrow0 * DHEAD + kdch]);
    async_ld_b128(&Ktile[0][ktrow1 * 64 + kdch], &Kp[(size_t)ktrow1 * DHEAD + kdch]);
    async_ld_b128(&Vtile[0][vdrow0 * 32 + vtch], &Vp[(size_t)vdrow0 * SLEN + vtch]);
    async_ld_b128(&Vtile[0][vdrow1 * 32 + vtch], &Vp[(size_t)vdrow1 * SLEN + vtch]);
  }

  for (int tb = 0; tb < SLEN; tb += 32) {
    const int buf = (tb >> 5) & 1;

    wait_async0();        // this wave's loads for `buf` complete
    __syncthreads();      // every wave has waited -> buf fully populated,
                          // and buf^1 is no longer being read by anyone

    // kick off next tile into the other buffer (overlaps with compute below)
    if (tb + 32 < SLEN) {
      const int nb = buf ^ 1;
      const int tn = tb + 32;
      async_ld_b128(&Ktile[nb][ktrow0 * 64 + kdch],
                    &Kp[(size_t)(tn + ktrow0) * DHEAD + kdch]);
      async_ld_b128(&Ktile[nb][ktrow1 * 64 + kdch],
                    &Kp[(size_t)(tn + ktrow1) * DHEAD + kdch]);
      async_ld_b128(&Vtile[nb][vdrow0 * 32 + vtch],
                    &Vp[(size_t)vdrow0 * SLEN + tn + vtch]);
      async_ld_b128(&Vtile[nb][vdrow1 * 32 + vtch],
                    &Vp[(size_t)vdrow1 * SLEN + tn + vtch]);
    }
    // warm L2 two tiles ahead
    if (tb + 64 < SLEN) {
      __builtin_prefetch(&Kp[(size_t)(tb + 64 + ktrow0) * DHEAD + kdch], 0, 1);
      __builtin_prefetch(&Vp[(size_t)vdrow0 * SLEN + tb + 64 + vtch], 0, 1);
    }

    // ---- scores: S = Q @ K^T  (two 16x16 tiles over 32 key columns) ----
    const bf16_t* Kt = Ktile[buf];
    const bf16_t* Vl = Vtile[buf];
    v8f s0 = {}, s1 = {};
    {
      v16bf bk;
      bk = load_b_frag_t(Kt, 64, 0, 0);    s0 = wmma_bf16(aq0, bk, s0);
      bk = load_b_frag_t(Kt, 64, 0, 32);   s0 = wmma_bf16(aq1, bk, s0);
      bk = load_b_frag_t(Kt, 64, 16, 0);   s1 = wmma_bf16(aq0, bk, s1);
      bk = load_b_frag_t(Kt, 64, 16, 32);  s1 = wmma_bf16(aq1, bk, s1);
    }

    // ---- online softmax over the 32 new columns ----
#pragma unroll
    for (int i = 0; i < 8; ++i) {
      float v0 = s0[i], v1 = s1[i];
      float mx = fmaxf(v0, v1);
#pragma unroll
      for (int off = 1; off < 16; off <<= 1)
        mx = fmaxf(mx, __shfl_xor(mx, off, 32));
      const float mnew  = fmaxf(mrow[i], mx);
      const float scale = __expf(mrow[i] - mnew);
      const float p0 = __expf(v0 - mnew);
      const float p1 = __expf(v1 - mnew);
      float rs = p0 + p1;
#pragma unroll
      for (int off = 1; off < 16; off <<= 1)
        rs += __shfl_xor(rs, off, 32);
      lrow[i] = lrow[i] * scale + rs;
      mrow[i] = mnew;
#pragma unroll
      for (int dt = 0; dt < 4; ++dt) oc[dt][i] *= scale;
      const int r = rowoff + i;
      pl[r * 32 + ncol]      = (bf16_t)p0;
      pl[r * 32 + 16 + ncol] = (bf16_t)p1;
    }

    // same-wave LDS RAW: DS ops are in-order; wait for stores before reads
    asm volatile("s_wait_dscnt 0" ::: "memory");

    // ---- O += P @ V  (contraction over the 32 keys) ----
    const v16bf ap = load_a_frag(pl, 32, 0, 0);
#pragma unroll
    for (int dt = 0; dt < 4; ++dt) {
      v16bf bv = load_b_frag_t(Vl, 32, dt * 16, 0);
      oc[dt] = wmma_bf16(ap, bv, oc[dt]);
    }
  }

  // ---- normalize and store O as [B,S,E] bf16 ----
  const int b = bh >> 4, h = bh & 15;
#pragma unroll
  for (int i = 0; i < 8; ++i) {
    const float inv = 1.0f / lrow[i];
    const int s = srow + rowoff + i;
    const size_t base = ((size_t)b * SLEN + s) * EMBED + h * DHEAD + ncol;
#pragma unroll
    for (int dt = 0; dt < 4; ++dt)
      Ob[base + dt * 16] = (bf16_t)(oc[dt][i] * inv);
  }
}

// ---------------------------------------------------------------------------
// Output projection: d_out[8192,1024]f32 = Ob @ Wproj + b_proj
// Same 32x64 wave tile as the QKV GEMM.
// ---------------------------------------------------------------------------
__global__ __launch_bounds__(256)
void proj_gemm_kernel(const bf16_t* __restrict__ Ab, const bf16_t* __restrict__ Wt,
                      const float* __restrict__ bias, float* __restrict__ out) {
  const int wave = threadIdx.x >> 5;
  const int lane = threadIdx.x & 31;
  const int M0 = blockIdx.x * 128 + (wave >> 1) * 32;
  const int N0 = blockIdx.y * 128 + (wave & 1) * 64;

  v8f acc[2][4] = {};
  for (int k0 = 0; k0 < EMBED; k0 += 32) {
    v16bf a0 = load_a_frag(Ab, EMBED, M0, k0);
    v16bf a1 = load_a_frag(Ab, EMBED, M0 + 16, k0);
#pragma unroll
    for (int ni = 0; ni < 4; ++ni) {
      v16bf b = load_b_frag_t(Wt, EMBED, N0 + ni * 16, k0);
      acc[0][ni] = wmma_bf16(a0, b, acc[0][ni]);
      acc[1][ni] = wmma_bf16(a1, b, acc[1][ni]);
    }
  }

  const int rowoff = (lane < 16) ? 0 : 8;
#pragma unroll
  for (int ni = 0; ni < 4; ++ni) {
    const int n = N0 + ni * 16 + (lane & 15);
    const float bv = bias[n];
#pragma unroll
    for (int mi = 0; mi < 2; ++mi) {
#pragma unroll
      for (int i = 0; i < 8; ++i)
        out[(size_t)(M0 + mi * 16 + rowoff + i) * EMBED + n] = acc[mi][ni][i] + bv;
    }
  }
}

// ---------------------------------------------------------------------------
// Launch
// ---------------------------------------------------------------------------
extern "C" void kernel_launch(void* const* d_in, const int* in_sizes, int n_in,
                              void* d_out, int out_size, void* d_ws, size_t ws_size,
                              hipStream_t stream) {
  const float* x      = (const float*)d_in[0];
  const float* w_qkv  = (const float*)d_in[1];
  const float* b_qkv  = (const float*)d_in[2];
  const float* w_proj = (const float*)d_in[3];
  const float* b_proj = (const float*)d_in[4];
  float* out = (float*)d_out;

  const size_t ROWS = (size_t)BATCH * SLEN;       // 8192
  const size_t szXb    = ROWS * EMBED * sizeof(bf16_t);
  const size_t szWqkvT = (size_t)3 * EMBED * EMBED * sizeof(bf16_t);
  const size_t szWprT  = (size_t)EMBED * EMBED * sizeof(bf16_t);
  const size_t szQ     = ROWS * EMBED * sizeof(bf16_t);
  const size_t need    = szXb + szWqkvT + szWprT + 4 * szQ;   // ~88 MB
  if (ws_size < need) return;

  char* ws = (char*)d_ws;
  bf16_t* Xb     = (bf16_t*)ws;            ws += szXb;
  bf16_t* WqkvT  = (bf16_t*)ws;            ws += szWqkvT;
  bf16_t* WprojT = (bf16_t*)ws;            ws += szWprT;
  bf16_t* Qb     = (bf16_t*)ws;            ws += szQ;
  bf16_t* Kb     = (bf16_t*)ws;            ws += szQ;
  bf16_t* Vt     = (bf16_t*)ws;            ws += szQ;
  bf16_t* Ob     = (bf16_t*)ws;            ws += szQ;

  // 1) convert inputs to bf16 (weights transposed for B-fragment loads)
  {
    int n = (int)(ROWS * EMBED);
    cvt_bf16_kernel<<<(n + 255) / 256, 256, 0, stream>>>(x, Xb, n);
    int nq = EMBED * 3 * EMBED;
    cvt_bf16_tr_kernel<3 * EMBED><<<(nq + 255) / 256, 256, 0, stream>>>(w_qkv, WqkvT, EMBED);
    int np = EMBED * EMBED;
    cvt_bf16_tr_kernel<EMBED><<<(np + 255) / 256, 256, 0, stream>>>(w_proj, WprojT, EMBED);
  }

  // 2) QKV projection (writes Q scaled, K, V-transposed)
  qkv_gemm_kernel<<<dim3(ROWS / 128, (3 * EMBED) / 128), 256, 0, stream>>>(
      Xb, WqkvT, b_qkv, Qb, Kb, Vt);

  // 3) flash attention: 64 (b,h) x 32 query blocks of 64 rows
  attn_kernel<<<BATCH * HEADS * (SLEN / 64), 128, 0, stream>>>(Qb, Kb, Vt, Ob);

  // 4) output projection -> fp32
  proj_gemm_kernel<<<dim3(ROWS / 128, EMBED / 128), 256, 0, stream>>>(
      Ob, WprojT, b_proj, out);
}
